// SharedExpertMoE_20796231647729
// MI455X (gfx1250) — matrix operand order repaired and checked by reading, not compiled, not run on hardware
//
#include <hip/hip_runtime.h>
#include <hip/hip_bf16.h>

typedef __attribute__((ext_vector_type(16))) __bf16 v16bf;
typedef __attribute__((ext_vector_type(8)))  __bf16 v8bf;
typedef __attribute__((ext_vector_type(8)))  float  v8f;
typedef __attribute__((ext_vector_type(4)))  float  v4f;

#define TOKENS 8192
#define DIMC   1024
#define HIDH   2048
#define NSH    2
#define NRT    6

// ---------------------------------------------------------------------------
// WMMA fragment loaders (16x16x32 bf16, wave32; contiguous k order used by this code for both A and B)
// A (16x32, 16-bit): lanes 0-15 -> row M=lane, K = k0+0..15 contiguous
//                    lanes 16-31 -> row M=lane-16, K = k0+16..31 contiguous
// B (32x16, 16-bit): lane n (n<16) -> col n, K = k0+0..15 contiguous
//                    lane n+16     -> col n, K = k0+16..31 contiguous
// C/D (16x16 f32):   VGPR r: lanes 0-15 -> M=r, N=lane; lanes 16-31 -> M=r+8
// ---------------------------------------------------------------------------
__device__ __forceinline__ v16bf load_a_frag(const __bf16* base, int ld, int m0,
                                             int k0, int lane) {
  int row = m0 + (lane & 15);
  int kk  = k0 + ((lane >> 4) << 4);            // +0 (low half) / +16 (high half)
  const __bf16* p = base + row * ld + kk;
  union { v16bf v; v8bf h[2]; } u;
  u.h[0] = *(const v8bf*)p;                     // K kk..kk+7
  u.h[1] = *(const v8bf*)(p + 8);               // K kk+8..kk+15
  return u.v;
}

__device__ __forceinline__ v16bf load_b_frag(const __bf16* wt, int ld, int n0,
                                             int k0, int lane) {
  int col = n0 + (lane & 15);
  int kk  = k0 + ((lane >> 4) << 4);            // +0 / +16
  return *(const v16bf*)(wt + (size_t)col * ld + kk);  // 32B contiguous
}

// ---------------------------------------------------------------------------
// Elementwise helpers
// ---------------------------------------------------------------------------
__global__ void zero_f32_kernel(float* __restrict__ p, int n4) {
  int i = blockIdx.x * blockDim.x + threadIdx.x;
  if (i < n4) ((v4f*)p)[i] = v4f{};
}

__global__ void f32_to_bf16_kernel(const float* __restrict__ in,
                                   __bf16* __restrict__ out, int n4) {
  int i = blockIdx.x * blockDim.x + threadIdx.x;
  if (i < n4) {
    v4f v = ((const v4f*)in)[i];
    __bf16* o = out + i * 4;
    o[0] = (__bf16)v.x; o[1] = (__bf16)v.y; o[2] = (__bf16)v.z; o[3] = (__bf16)v.w;
  }
}

// Transposed fp32 -> bf16 cast: in [R,C] row-major -> out [C,R] row-major.
__global__ void transpose_to_bf16_kernel(const float* __restrict__ in,
                                         __bf16* __restrict__ out, int R, int C) {
  __shared__ float tile[32][33];
  int c0 = blockIdx.x * 32, r0 = blockIdx.y * 32;
  int tx = threadIdx.x & 31, ty = threadIdx.x >> 5;   // 256 thr: ty 0..7
  for (int i = ty; i < 32; i += 8)
    tile[i][tx] = in[(size_t)(r0 + i) * C + c0 + tx];
  __syncthreads();
  for (int i = ty; i < 32; i += 8)
    out[(size_t)(c0 + i) * R + r0 + tx] = (__bf16)tile[tx][i];
}

// ---------------------------------------------------------------------------
// Router: one wave per token. logits -> top2 softmax weights -> expert lists,
// plus aux-loss statistics (top-1 counts, mean full-softmax probs).
// ---------------------------------------------------------------------------
__global__ void gate_kernel(const float* __restrict__ x,
                            const float* __restrict__ gw,
                            int* __restrict__ tok_list, float* __restrict__ wt_list,
                            int* __restrict__ counts, int* __restrict__ f_cnt,
                            float* __restrict__ p_sum) {
  int tok  = (blockIdx.x * blockDim.x + threadIdx.x) >> 5;
  int lane = threadIdx.x & 31;
  if (tok >= TOKENS) return;
  const float* xr = x + (size_t)tok * DIMC;
  float acc[NRT] = {};
  for (int k = lane; k < DIMC; k += 32) {
    float xv = xr[k];
#pragma unroll
    for (int e = 0; e < NRT; e++) acc[e] += xv * gw[k * NRT + e];
  }
#pragma unroll
  for (int e = 0; e < NRT; e++)
    for (int off = 16; off > 0; off >>= 1) acc[e] += __shfl_xor(acc[e], off, 32);
  if (lane == 0) {
    int i0 = 0;
    for (int e = 1; e < NRT; e++) if (acc[e] > acc[i0]) i0 = e;   // first max
    int i1 = -1;
    for (int e = 0; e < NRT; e++)
      if (e != i0 && (i1 < 0 || acc[e] > acc[i1])) i1 = e;
    float e1 = __expf(acc[i1] - acc[i0]);                         // acc[i0] is max
    float inv = 1.0f / (1.0f + e1);
    float w0 = inv, w1 = e1 * inv;
    int s0 = atomicAdd(&counts[i0], 1);
    tok_list[i0 * TOKENS + s0] = tok; wt_list[i0 * TOKENS + s0] = w0;
    int s1 = atomicAdd(&counts[i1], 1);
    tok_list[i1 * TOKENS + s1] = tok; wt_list[i1 * TOKENS + s1] = w1;
    atomicAdd(&f_cnt[i0], 1);
    float mx = acc[0];
    for (int e = 1; e < NRT; e++) mx = fmaxf(mx, acc[e]);
    float pe[NRT], se = 0.0f;
    for (int e = 0; e < NRT; e++) { pe[e] = __expf(acc[e] - mx); se += pe[e]; }
    float isе = 1.0f / se;
    for (int e = 0; e < NRT; e++) atomicAdd(&p_sum[e], pe[e] * isе);
  }
}

// ---------------------------------------------------------------------------
// Fused SwiGLU expert pass. One WG owns a 32-token tile:
//   LDS: x-tile 32x1024 bf16 (64KB) + h-tile 32x2048 bf16 (128KB) + row meta.
//   GEMM1: h = silu(x@W1) * (x@W3)  (W1t/W3t are [H,K] bf16, K-contiguous)
//   GEMM2: combined[tok] += w_row * (h @ W2)  via global f32 atomics.
// tok_list == nullptr -> shared expert (identity rows, weight 1).
// ---------------------------------------------------------------------------
__global__ __launch_bounds__(256)
void ffn_kernel(const __bf16* __restrict__ xb,
                const __bf16* __restrict__ w1t,   // [HIDH, DIMC]
                const __bf16* __restrict__ w3t,   // [HIDH, DIMC]
                const __bf16* __restrict__ w2t,   // [DIMC, HIDH]
                float* __restrict__ combined,     // [TOKENS, DIMC]
                const int* __restrict__ tok_list,
                const float* __restrict__ wt_list,
                const int* __restrict__ count_ptr) {
  extern __shared__ char smem[];
  __bf16* a_lds   = (__bf16*)smem;                              // 32 x 1024
  __bf16* h_lds   = (__bf16*)(smem + 32 * DIMC * 2);            // 32 x 2048
  int*    tok_lds = (int*)  (smem + 32 * DIMC * 2 + 32 * HIDH * 2);
  float*  w_lds   = (float*)(smem + 32 * DIMC * 2 + 32 * HIDH * 2 + 32 * 4);

  int tid = threadIdx.x, lane = tid & 31, wid = tid >> 5;
  int cnt = count_ptr ? *count_ptr : TOKENS;
  if ((int)blockIdx.x * 32 >= cnt) return;        // uniform exit, EXEC stays full

  if (tid < 32) {
    int rg = blockIdx.x * 32 + tid;
    int t; float w;
    if (tok_list) {
      if (rg < cnt) { t = tok_list[rg]; w = wt_list[rg]; }
      else          { t = 0;            w = 0.0f; }   // padded row: zero weight
    } else { t = rg; w = 1.0f; }
    tok_lds[tid] = t; w_lds[tid] = w;
  }
  __syncthreads();

  for (int idx = tid; idx < 32 * DIMC / 8; idx += 256) {        // gather x-tile
    int row = idx >> 7, off = (idx & 127) << 3;
    *(v8bf*)(a_lds + row * DIMC + off) =
        *(const v8bf*)(xb + (size_t)tok_lds[row] * DIMC + off);
  }
  __syncthreads();

  int m0  = (wid & 1) << 4;       // 2 M-blocks of 16 rows
  int grp = wid >> 1;             // 4 N-groups

  // ---- GEMM1 + SwiGLU -> h_lds ----
  for (int n0 = grp * 512; n0 < grp * 512 + 512; n0 += 16) {
    v8f acc1 = {}, acc3 = {};
    for (int k0 = 0; k0 < DIMC; k0 += 32) {
      v16bf a  = load_a_frag(a_lds, DIMC, m0, k0, lane);
      v16bf b1 = load_b_frag(w1t, DIMC, n0, k0, lane);
      v16bf b3 = load_b_frag(w3t, DIMC, n0, k0, lane);
      acc1 = __builtin_amdgcn_wmma_f32_16x16x32_bf16(false, a, false, b1,
                                                     (short)0, acc1, false, false);
      acc3 = __builtin_amdgcn_wmma_f32_16x16x32_bf16(false, a, false, b3,
                                                     (short)0, acc3, false, false);
    }
    int col = n0 + (lane & 15);
    int rof = m0 + ((lane >> 4) << 3);
#pragma unroll
    for (int i = 0; i < 8; i++) {
      float g = acc1[i];
      float h = (g / (1.0f + __expf(-g))) * acc3[i];   // silu(g) * v
      h_lds[(rof + i) * HIDH + col] = (__bf16)h;
    }
  }
  __syncthreads();

  // ---- GEMM2: combined += w_row * (h @ W2) ----
  for (int c0 = grp * 256; c0 < grp * 256 + 256; c0 += 16) {
    v8f acc = {};
    for (int k0 = 0; k0 < HIDH; k0 += 32) {
      v16bf a = load_a_frag(h_lds, HIDH, m0, k0, lane);
      v16bf b = load_b_frag(w2t, HIDH, c0, k0, lane);
      acc = __builtin_amdgcn_wmma_f32_16x16x32_bf16(false, a, false, b,
                                                    (short)0, acc, false, false);
    }
    int col = c0 + (lane & 15);
    int rof = m0 + ((lane >> 4) << 3);
#pragma unroll
    for (int i = 0; i < 8; i++) {
      int r = rof + i;
      atomicAdd(&combined[(size_t)tok_lds[r] * DIMC + col], acc[i] * w_lds[r]);
    }
  }
}

// ---------------------------------------------------------------------------
// Final projection: out = combined @ out_w  (plain stores, f32 output)
// ---------------------------------------------------------------------------
__global__ __launch_bounds__(256)
void out_gemm_kernel(const __bf16* __restrict__ ab,     // [TOKENS, DIMC] bf16
                     const __bf16* __restrict__ owt,    // [DIMC, DIMC] (N,K)
                     float* __restrict__ out) {
  extern __shared__ char smem[];
  __bf16* a_lds = (__bf16*)smem;                        // 32 x 1024
  int tid = threadIdx.x, lane = tid & 31, wid = tid >> 5;
  int row0 = blockIdx.x * 32;
  for (int idx = tid; idx < 32 * DIMC / 8; idx += 256) {
    int row = idx >> 7, off = (idx & 127) << 3;
    *(v8bf*)(a_lds + row * DIMC + off) =
        *(const v8bf*)(ab + (size_t)(row0 + row) * DIMC + off);
  }
  __syncthreads();
  int m0 = (wid & 1) << 4, grp = wid >> 1;
  for (int c0 = grp * 256; c0 < grp * 256 + 256; c0 += 16) {
    v8f acc = {};
    for (int k0 = 0; k0 < DIMC; k0 += 32) {
      v16bf a = load_a_frag(a_lds, DIMC, m0, k0, lane);
      v16bf b = load_b_frag(owt, DIMC, c0, k0, lane);
      acc = __builtin_amdgcn_wmma_f32_16x16x32_bf16(false, a, false, b,
                                                    (short)0, acc, false, false);
    }
    int col = c0 + (lane & 15);
    int rof = m0 + ((lane >> 4) << 3);
#pragma unroll
    for (int i = 0; i < 8; i++)
      out[(size_t)(row0 + rof + i) * DIMC + col] = acc[i];
  }
}

__global__ void aux_kernel(const int* __restrict__ f_cnt,
                           const float* __restrict__ p_sum,
                           float* __restrict__ out_aux) {
  if (threadIdx.x == 0 && blockIdx.x == 0) {
    float s = 0.0f;
    for (int e = 0; e < NRT; e++)
      s += ((float)f_cnt[e] / (float)TOKENS) * (p_sum[e] / (float)TOKENS);
    *out_aux = 0.01f * (float)NRT * s;
  }
}

// ---------------------------------------------------------------------------
extern "C" void kernel_launch(void* const* d_in, const int* in_sizes, int n_in,
                              void* d_out, int out_size, void* d_ws, size_t ws_size,
                              hipStream_t stream) {
  (void)in_sizes; (void)n_in; (void)out_size; (void)ws_size;
  const float* x    = (const float*)d_in[0];
  const float* w1_s = (const float*)d_in[1];
  const float* w2_s = (const float*)d_in[2];
  const float* w3_s = (const float*)d_in[3];
  const float* w1_r = (const float*)d_in[4];
  const float* w2_r = (const float*)d_in[5];
  const float* w3_r = (const float*)d_in[6];
  const float* gw   = (const float*)d_in[7];
  const float* ow   = (const float*)d_in[8];
  float* outf = (float*)d_out;

  // ---- workspace carve-up (bytes) ----
  char* ws = (char*)d_ws;
  size_t off = 0;
  const size_t WSZ = (size_t)DIMC * HIDH * 2;               // one weight matrix, bf16
  __bf16* xb     = (__bf16*)(ws + off); off += (size_t)TOKENS * DIMC * 2;
  __bf16* w1t_s  = (__bf16*)(ws + off); off += NSH * WSZ;
  __bf16* w3t_s  = (__bf16*)(ws + off); off += NSH * WSZ;
  __bf16* w2t_s  = (__bf16*)(ws + off); off += NSH * WSZ;
  __bf16* w1t_r  = (__bf16*)(ws + off); off += NRT * WSZ;
  __bf16* w3t_r  = (__bf16*)(ws + off); off += NRT * WSZ;
  __bf16* w2t_r  = (__bf16*)(ws + off); off += NRT * WSZ;
  __bf16* outwt  = (__bf16*)(ws + off); off += (size_t)DIMC * DIMC * 2;
  float*  comb   = (float*)(ws + off);  off += (size_t)TOKENS * DIMC * 4;
  __bf16* combb  = (__bf16*)(ws + off); off += (size_t)TOKENS * DIMC * 2;
  int*    tokls  = (int*)(ws + off);    off += (size_t)NRT * TOKENS * 4;
  float*  wtls   = (float*)(ws + off);  off += (size_t)NRT * TOKENS * 4;
  float*  ctrl   = (float*)(ws + off);  off += 256;  // counts[6] | f_cnt[6] | p_sum[6]
  int*   counts = (int*)ctrl;
  int*   f_cnt  = (int*)ctrl + NRT;
  float* p_sum  = ctrl + 2 * NRT;

  const size_t NTOT = (size_t)TOKENS * DIMC;                // 8.39M
  const int FFN_LDS = 32 * DIMC * 2 + 32 * HIDH * 2 + 32 * 8;  // 196,864 B
  const int OUT_LDS = 32 * DIMC * 2;                           // 65,536 B
  (void)hipFuncSetAttribute((const void*)ffn_kernel,
      hipFuncAttributeMaxDynamicSharedMemorySize, FFN_LDS);
  (void)hipFuncSetAttribute((const void*)out_gemm_kernel,
      hipFuncAttributeMaxDynamicSharedMemorySize, OUT_LDS);

  // 1) zero accumulator + control block
  zero_f32_kernel<<<(int)(NTOT / 4 + 255) / 256, 256, 0, stream>>>(comb, (int)(NTOT / 4));
  zero_f32_kernel<<<1, 64, 0, stream>>>(ctrl, 64);

  // 2) cast x to bf16; cast+transpose all weights to bf16 [N,K]
  f32_to_bf16_kernel<<<(int)(NTOT / 4 + 255) / 256, 256, 0, stream>>>(x, xb, (int)(NTOT / 4));
  const size_t WE = (size_t)DIMC * HIDH;                    // elements per matrix
  for (int e = 0; e < NSH; e++) {
    transpose_to_bf16_kernel<<<dim3(HIDH / 32, DIMC / 32), 256, 0, stream>>>(
        w1_s + e * WE, w1t_s + e * WE, DIMC, HIDH);
    transpose_to_bf16_kernel<<<dim3(HIDH / 32, DIMC / 32), 256, 0, stream>>>(
        w3_s + e * WE, w3t_s + e * WE, DIMC, HIDH);
    transpose_to_bf16_kernel<<<dim3(DIMC / 32, HIDH / 32), 256, 0, stream>>>(
        w2_s + e * WE, w2t_s + e * WE, HIDH, DIMC);
  }
  for (int e = 0; e < NRT; e++) {
    transpose_to_bf16_kernel<<<dim3(HIDH / 32, DIMC / 32), 256, 0, stream>>>(
        w1_r + e * WE, w1t_r + e * WE, DIMC, HIDH);
    transpose_to_bf16_kernel<<<dim3(HIDH / 32, DIMC / 32), 256, 0, stream>>>(
        w3_r + e * WE, w3t_r + e * WE, DIMC, HIDH);
    transpose_to_bf16_kernel<<<dim3(DIMC / 32, HIDH / 32), 256, 0, stream>>>(
        w2_r + e * WE, w2t_r + e * WE, HIDH, DIMC);
  }
  transpose_to_bf16_kernel<<<dim3(DIMC / 32, DIMC / 32), 256, 0, stream>>>(
      ow, outwt, DIMC, DIMC);

  // 3) router + expert token lists + aux stats
  gate_kernel<<<TOKENS * 32 / 256, 256, 0, stream>>>(x, gw, tokls, wtls,
                                                     counts, f_cnt, p_sum);

  // 4) expert passes: 2 shared (all tokens) + 6 routed (gathered top-2 lists)
  for (int e = 0; e < NSH; e++)
    ffn_kernel<<<TOKENS / 32, 256, FFN_LDS, stream>>>(
        xb, w1t_s + e * WE, w3t_s + e * WE, w2t_s + e * WE, comb,
        nullptr, nullptr, nullptr);
  for (int e = 0; e < NRT; e++)
    ffn_kernel<<<TOKENS / 32, 256, FFN_LDS, stream>>>(
        xb, w1t_r + e * WE, w3t_r + e * WE, w2t_r + e * WE, comb,
        tokls + e * TOKENS, wtls + e * TOKENS, counts + e);

  // 5) final projection + aux loss
  f32_to_bf16_kernel<<<(int)(NTOT / 4 + 255) / 256, 256, 0, stream>>>(comb, combb, (int)(NTOT / 4));
  out_gemm_kernel<<<TOKENS / 32, 256, OUT_LDS, stream>>>(combb, outwt, outf);
  aux_kernel<<<1, 32, 0, stream>>>(f_cnt, p_sum, outf + NTOT);
}
